// ProtoDINO_78967268704261
// MI455X (gfx1250) — matrix operand order repaired and checked by reading, not compiled, not run on hardware
//
#include <hip/hip_runtime.h>

// ---------------- problem constants ----------------
#define BB    32
#define NP    1369
#define DIM   768
#define NC    201          // classes incl. background
#define KP    5
#define COLS  1005         // NC*KP
#define ROWS  43808        // BB*NP
#define NCLS  200
#define INV_EPS 20.0f      // 1/SK_EPS

typedef __attribute__((ext_vector_type(16))) __bf16 v16bf;
typedef __attribute__((ext_vector_type(8)))  float  v8f;

union Frag { v16bf v; uint4 q[2]; };

__device__ __forceinline__ v8f wmma_bf16(const Frag& a, const Frag& b, v8f c) {
    return __builtin_amdgcn_wmma_f32_16x16x32_bf16(false, a.v, false, b.v,
                                                   (short)0, c, false, false);
}

__device__ __forceinline__ unsigned short f2bf(float f) {
    unsigned u = __float_as_uint(f);
    unsigned r = u + 0x7fffu + ((u >> 16) & 1u);   // round-to-nearest-even
    return (unsigned short)(r >> 16);
}
__device__ __forceinline__ float bf2f(unsigned short h) {
    return __uint_as_float(((unsigned)h) << 16);
}
__device__ __forceinline__ unsigned fkey(float f) {      // order-preserving uint key
    unsigned u = __float_as_uint(f);
    return (u & 0x80000000u) ? ~u : (u | 0x80000000u);
}
__device__ __forceinline__ float fdec(unsigned k) {
    return (k & 0x80000000u) ? __uint_as_float(k ^ 0x80000000u) : __uint_as_float(~k);
}

// ---------------- init / zero workspace ----------------
__global__ void k_init(float* Pacc, float* S, float* mean,
                       int* present, int* ncnt, unsigned* mmkey) {
    int i = blockIdx.x * 256 + threadIdx.x;
    if (i < NC * KP * DIM) Pacc[i] = 0.f;
    if (i < COLS)          S[i] = 0.f;
    if (i < DIM)           mean[i] = 0.f;
    if (i < NC)            { present[i] = 0; ncnt[i] = 0; }
    if (i == 0)            { mmkey[0] = 0xFFFFFFFFu; mmkey[1] = 0u; }
}

// ---------------- transpose X (f32 row-major) -> Xt (bf16, 768 x 43808) ----------------
__global__ __launch_bounds__(256) void k_transpose(const float* __restrict__ X,
                                                   unsigned short* __restrict__ Xt) {
    __shared__ unsigned short tile[32][33];
    int bx = blockIdx.x % 1369, by = blockIdx.x / 1369;
    int r0 = bx * 32, d0 = by * 32, t = threadIdx.x;
    for (int rep = 0; rep < 4; ++rep) {
        int idx = t + rep * 256, r = idx >> 5, d = idx & 31;
        tile[r][d] = f2bf(X[(size_t)(r0 + r) * DIM + d0 + d]);
    }
    __syncthreads();
    for (int rep = 0; rep < 4; ++rep) {
        int idx = t + rep * 256, d = idx >> 5, r = idx & 31;
        Xt[(size_t)(d0 + d) * ROWS + (r0 + r)] = tile[r][d];
    }
}

// ---------------- Gram = X^T X (768x768, f32 acc) via bf16 WMMA ----------------
__global__ __launch_bounds__(256) void k_gram(const unsigned short* __restrict__ Xt,
                                              float* __restrict__ G) {
    const int LD = ROWS / 8;                      // uint4 per Xt row
    int tid = threadIdx.x, wave = tid >> 5, lane = tid & 31;
    int half = lane >> 4, lm = lane & 15;
    int task = blockIdx.x * 8 + wave;             // 288 tasks = 24 x 12
    int rowT = task / 12, colG = task % 12;
    int r0 = rowT * 32, c0 = colG * 64;
    const uint4* Xq = (const uint4*)Xt;
    int rA0 = (r0 + lm) * LD, rA1 = (r0 + 16 + lm) * LD;
    int cb[4];
    for (int j = 0; j < 4; ++j) cb[j] = (c0 + 16 * j + lm) * LD;
    v8f z = {0.f,0.f,0.f,0.f,0.f,0.f,0.f,0.f};
    v8f acc[2][4];
    for (int i = 0; i < 2; ++i) for (int j = 0; j < 4; ++j) acc[i][j] = z;
#pragma unroll 2
    for (int k0 = 0; k0 < ROWS; k0 += 32) {
        int ka8 = (k0 + 16 * half) >> 3;          // uint4 index of A chunk 0
        int kb8 = (k0 + 16 * half) >> 3;          // uint4 index of B chunk 0
        Frag a0, a1, b0, b1, b2, b3;
        a0.q[0] = Xq[rA0 + ka8]; a0.q[1] = Xq[rA0 + ka8 + 1];
        a1.q[0] = Xq[rA1 + ka8]; a1.q[1] = Xq[rA1 + ka8 + 1];
        b0.q[0] = Xq[cb[0] + kb8]; b0.q[1] = Xq[cb[0] + kb8 + 1];
        b1.q[0] = Xq[cb[1] + kb8]; b1.q[1] = Xq[cb[1] + kb8 + 1];
        b2.q[0] = Xq[cb[2] + kb8]; b2.q[1] = Xq[cb[2] + kb8 + 1];
        b3.q[0] = Xq[cb[3] + kb8]; b3.q[1] = Xq[cb[3] + kb8 + 1];
        acc[0][0] = wmma_bf16(a0, b0, acc[0][0]);
        acc[0][1] = wmma_bf16(a0, b1, acc[0][1]);
        acc[0][2] = wmma_bf16(a0, b2, acc[0][2]);
        acc[0][3] = wmma_bf16(a0, b3, acc[0][3]);
        acc[1][0] = wmma_bf16(a1, b0, acc[1][0]);
        acc[1][1] = wmma_bf16(a1, b1, acc[1][1]);
        acc[1][2] = wmma_bf16(a1, b2, acc[1][2]);
        acc[1][3] = wmma_bf16(a1, b3, acc[1][3]);
    }
    for (int mi = 0; mi < 2; ++mi)
        for (int j = 0; j < 4; ++j) {
            int col = c0 + 16 * j + lm;
            int rb = r0 + 16 * mi + 8 * half;
            for (int jj = 0; jj < 8; ++jj)
                G[(size_t)(rb + jj) * DIM + col] = acc[mi][j][jj];
        }
}

// ---------------- column mean of X ----------------
__global__ __launch_bounds__(256) void k_mean(const float* __restrict__ X,
                                              float* __restrict__ mean) {
    int b = blockIdx.x, t = threadIdx.x;
    int i0 = b * 685, i1 = min(i0 + 685, ROWS);
    float a0 = 0.f, a1 = 0.f, a2 = 0.f;
    for (int i = i0; i < i1; ++i) {
        const float* row = X + (size_t)i * DIM;
        a0 += row[t]; a1 += row[t + 256]; a2 += row[t + 512];
    }
    const float inv = 1.0f / (float)ROWS;
    atomicAdd(&mean[t],       a0 * inv);
    atomicAdd(&mean[t + 256], a1 * inv);
    atomicAdd(&mean[t + 512], a2 * inv);
}

// ---------------- power iteration on centered Gram: Gc v = G v - N m (m.v) ----------------
__global__ __launch_bounds__(768) void k_power(const float* __restrict__ G,
                                               const float* __restrict__ mean,
                                               float* __restrict__ vout,
                                               float* __restrict__ mvout) {
    __shared__ float vs[768];
    __shared__ float red[1024];
    int t = threadIdx.x;
    unsigned h = t * 747796405u + 2891336453u;    // deterministic pseudo-random init
    h ^= h >> 16; h *= 2246822519u; h ^= h >> 13;
    vs[t] = ((float)(h & 0xFFFFu) / 65535.0f) - 0.5f + 1e-3f;
    if (t < 256) red[768 + t] = 0.f;
    __syncthreads();
    float m_t = mean[t];
    for (int it = 0; it < 10; ++it) {
        red[t] = m_t * vs[t]; __syncthreads();
        for (int s = 512; s > 0; s >>= 1) { if (t < s) red[t] += red[t + s]; __syncthreads(); }
        float mv = red[0]; __syncthreads();
        float s = 0.f;
        const float* gr = G + (size_t)t * DIM;
        for (int d = 0; d < DIM; ++d) s += gr[d] * vs[d];
        s -= (float)ROWS * m_t * mv;
        red[t] = s * s; __syncthreads();
        for (int s2 = 512; s2 > 0; s2 >>= 1) { if (t < s2) red[t] += red[t + s2]; __syncthreads(); }
        float nrm = sqrtf(red[0]); __syncthreads();
        vs[t] = s / (nrm + 1e-12f); __syncthreads();
    }
    vout[t] = vs[t];
    red[t] = m_t * vs[t]; __syncthreads();
    for (int s = 512; s > 0; s >>= 1) { if (t < s) red[t] += red[t + s]; __syncthreads(); }
    if (t == 0) mvout[0] = red[0];
}

// ---------------- u = Xc v (one wave per row) + global min/max ----------------
__global__ __launch_bounds__(256) void k_u(const float* __restrict__ X,
                                           const float* __restrict__ v,
                                           const float* __restrict__ mv,
                                           float* __restrict__ u, unsigned* mmkey) {
    int tid = threadIdx.x, lane = tid & 31, wave = tid >> 5;
    int i = blockIdx.x * 8 + wave;
    if (i >= ROWS) return;
    const float* row = X + (size_t)i * DIM;
    float s = 0.f;
    for (int d = lane; d < DIM; d += 32) s += row[d] * v[d];
    for (int o = 16; o; o >>= 1) s += __shfl_xor(s, o, 32);
    if (lane == 0) {
        float ui = s - mv[0];
        u[i] = ui;
        unsigned k = fkey(ui);
        atomicMin(&mmkey[0], k);
        atomicMax(&mmkey[1], k);
    }
}

// ---------------- pseudo labels, mask r-init, fg counts, present flags ----------------
__global__ __launch_bounds__(256) void k_pseudo(const float* __restrict__ u,
                                                const unsigned* __restrict__ mmkey,
                                                const int* __restrict__ labels,
                                                int* __restrict__ clsbuf,
                                                float* __restrict__ o_pseudo,
                                                float* __restrict__ rbuf,
                                                int* __restrict__ ncnt,
                                                int* __restrict__ present) {
    __shared__ int cnt[256];
    int b = blockIdx.x, t = threadIdx.x;
    float umin = fdec(mmkey[0]), umax = fdec(mmkey[1]);
    float inv = 1.0f / fmaxf(umax - umin, 1e-20f);
    int lbl = labels[b], c = 0;
    for (int n = t; n < NP; n += 256) {
        int i = b * NP + n;
        float us = (u[i] - umin) * inv;
        bool fg = us < 0.5f;
        int cl = fg ? lbl : (NC - 1);
        clsbuf[i] = cl;
        o_pseudo[i] = (float)cl;
        rbuf[i] = fg ? 1.0f : 0.0f;
        c += fg ? 1 : 0;
    }
    cnt[t] = c; __syncthreads();
    for (int s = 128; s > 0; s >>= 1) { if (t < s) cnt[t] += cnt[t + s]; __syncthreads(); }
    if (t == 0) { atomicAdd(&ncnt[lbl], cnt[0]); present[lbl] = 1; }
}

// ---------------- l2-normalize rows -> bf16 (tokens / prototypes) ----------------
__global__ __launch_bounds__(256) void k_norm_tokens(const float* __restrict__ X,
                                                     unsigned short* __restrict__ Xn) {
    int tid = threadIdx.x, lane = tid & 31, wave = tid >> 5;
    int i = blockIdx.x * 8 + wave;
    if (i >= ROWS) return;
    const float* row = X + (size_t)i * DIM;
    float xr[24], ss = 0.f;
    for (int j = 0; j < 24; ++j) { float x = row[lane + j * 32]; xr[j] = x; ss += x * x; }
    for (int o = 16; o; o >>= 1) ss += __shfl_xor(ss, o, 32);
    float inv = 1.0f / fmaxf(sqrtf(ss), 1e-12f);
    for (int j = 0; j < 24; ++j) Xn[(size_t)i * DIM + lane + j * 32] = f2bf(xr[j] * inv);
}
__global__ __launch_bounds__(256) void k_norm_proto(const float* __restrict__ P,
                                                    unsigned short* __restrict__ Pn) {
    int tid = threadIdx.x, lane = tid & 31, wave = tid >> 5;
    int i = blockIdx.x * 8 + wave;
    if (i >= COLS) return;
    const float* row = P + (size_t)i * DIM;
    float xr[24], ss = 0.f;
    for (int j = 0; j < 24; ++j) { float x = row[lane + j * 32]; xr[j] = x; ss += x * x; }
    for (int o = 16; o; o >>= 1) ss += __shfl_xor(ss, o, 32);
    float inv = 1.0f / fmaxf(sqrtf(ss), 1e-12f);
    for (int j = 0; j < 24; ++j) Pn[(size_t)i * DIM + lane + j * 32] = f2bf(xr[j] * inv);
}

// ---------------- main GEMM: logits = Xn (43808x768) * Pn^T (768x1005), bf16 WMMA ----------------
__global__ __launch_bounds__(256) void k_gemm_logits(const unsigned short* __restrict__ Xn,
                                                     const unsigned short* __restrict__ Pn,
                                                     float* __restrict__ out) {
    int tid = threadIdx.x, wave = tid >> 5, lane = tid & 31;
    int half = lane >> 4, lm = lane & 15;
    int task = blockIdx.x * 8 + wave;             // 21904 = 1369 rowT32 x 16 colG64
    int rowT = task >> 4, colG = task & 15;
    int r0 = rowT * 32, c0 = colG * 64;
    const uint4* Xq = (const uint4*)Xn;           // 96 uint4 per 768-dim row
    const uint4* Pq = (const uint4*)Pn;
    int rA0 = (r0 + lm) * 96, rA1 = (r0 + 16 + lm) * 96;
    int cb[4], cols[4];
    for (int j = 0; j < 4; ++j) {
        int col = c0 + 16 * j + lm;
        cols[j] = col;
        cb[j] = (col < COLS ? col : COLS - 1) * 96;   // clamp OOB reads
    }
    v8f z = {0.f,0.f,0.f,0.f,0.f,0.f,0.f,0.f};
    v8f acc[2][4];
    for (int i = 0; i < 2; ++i) for (int j = 0; j < 4; ++j) acc[i][j] = z;
#pragma unroll 4
    for (int k0 = 0; k0 < DIM; k0 += 32) {
        int ka8 = (k0 + 16 * half) >> 3;
        int kb8 = (k0 + 16 * half) >> 3;
        Frag a0, a1, b0, b1, b2, b3;
        a0.q[0] = Xq[rA0 + ka8]; a0.q[1] = Xq[rA0 + ka8 + 1];
        a1.q[0] = Xq[rA1 + ka8]; a1.q[1] = Xq[rA1 + ka8 + 1];
        b0.q[0] = Pq[cb[0] + kb8]; b0.q[1] = Pq[cb[0] + kb8 + 1];
        b1.q[0] = Pq[cb[1] + kb8]; b1.q[1] = Pq[cb[1] + kb8 + 1];
        b2.q[0] = Pq[cb[2] + kb8]; b2.q[1] = Pq[cb[2] + kb8 + 1];
        b3.q[0] = Pq[cb[3] + kb8]; b3.q[1] = Pq[cb[3] + kb8 + 1];
        acc[0][0] = wmma_bf16(a0, b0, acc[0][0]);
        acc[0][1] = wmma_bf16(a0, b1, acc[0][1]);
        acc[0][2] = wmma_bf16(a0, b2, acc[0][2]);
        acc[0][3] = wmma_bf16(a0, b3, acc[0][3]);
        acc[1][0] = wmma_bf16(a1, b0, acc[1][0]);
        acc[1][1] = wmma_bf16(a1, b1, acc[1][1]);
        acc[1][2] = wmma_bf16(a1, b2, acc[1][2]);
        acc[1][3] = wmma_bf16(a1, b3, acc[1][3]);
    }
    for (int mi = 0; mi < 2; ++mi)
        for (int j = 0; j < 4; ++j) {
            if (cols[j] < COLS) {
                int rb = r0 + 16 * mi + 8 * half;
                for (int jj = 0; jj < 8; ++jj)
                    out[(size_t)(rb + jj) * COLS + cols[j]] = acc[mi][j][jj];
            }
        }
}

// ---------------- sinkhorn (scaling-factor form Q = r_i c_j e_ij) ----------------
__global__ __launch_bounds__(256) void k_skS(const float* __restrict__ logits,
                                             const int* __restrict__ labels,
                                             const float* __restrict__ rbuf,
                                             float* __restrict__ S) {
    __shared__ float sred[256];
    int b = blockIdx.x, t = threadIdx.x, lbl = labels[b];
    float acc[5] = {0.f, 0.f, 0.f, 0.f, 0.f};
    for (int n = t; n < NP; n += 256) {
        int i = b * NP + n;
        float r = rbuf[i];
        if (r > 0.f) {
            const float* L = logits + (size_t)i * COLS + lbl * KP;
            for (int j = 0; j < KP; ++j) acc[j] += r * expf(L[j] * INV_EPS);
        }
    }
    for (int j = 0; j < KP; ++j) {
        sred[t] = acc[j]; __syncthreads();
        for (int s = 128; s > 0; s >>= 1) { if (t < s) sred[t] += sred[t + s]; __syncthreads(); }
        if (t == 0) atomicAdd(&S[lbl * KP + j], sred[0]);
        __syncthreads();
    }
}
__global__ void k_skC(float* S, float* cvec) {
    int i = blockIdx.x * 256 + threadIdx.x;
    if (i >= COLS) return;
    cvec[i] = 1.0f / ((float)KP * fmaxf(S[i], 1e-12f));
    S[i] = 0.f;                                   // ready for next iteration
}
__global__ void k_skR(const float* __restrict__ logits, const int* __restrict__ clsbuf,
                      const float* __restrict__ cvec, const int* __restrict__ ncnt,
                      float* __restrict__ rbuf) {
    int i = blockIdx.x * 256 + threadIdx.x;
    if (i >= ROWS) return;
    int c = clsbuf[i];
    if (c >= NCLS) return;                        // background: rbuf stays 0
    const float* L = logits + (size_t)i * COLS + c * KP;
    float den = 0.f;
    for (int j = 0; j < KP; ++j) den += cvec[c * KP + j] * expf(L[j] * INV_EPS);
    float nc = (float)max(ncnt[c], 1);
    rbuf[i] = 1.0f / (nc * fmaxf(den, 1e-12f));
}

// ---------------- P_new accumulation: Pacc[c] += A^T Xn, per-image blocks ----------------
__global__ __launch_bounds__(768) void k_paccum(const float* __restrict__ logits,
                                                const unsigned short* __restrict__ Xn,
                                                const int* __restrict__ labels,
                                                const float* __restrict__ rbuf,
                                                const float* __restrict__ cvec,
                                                const int* __restrict__ ncnt,
                                                float* __restrict__ Pacc) {
    __shared__ float Ash[64 * KP];
    int b = blockIdx.x, t = threadIdx.x, lbl = labels[b];
    float nc = (float)max(ncnt[lbl], 1);
    float cl[KP];
    for (int j = 0; j < KP; ++j) cl[j] = cvec[lbl * KP + j];
    float acc[KP] = {0.f, 0.f, 0.f, 0.f, 0.f};
    for (int base = 0; base < NP; base += 64) {
        __syncthreads();
        if (t < 64) {
            int n = base + t;
            float a[KP] = {0.f, 0.f, 0.f, 0.f, 0.f};
            if (n < NP) {
                int i = b * NP + n;
                float r = rbuf[i];
                if (r > 0.f) {
                    const float* L = logits + (size_t)i * COLS + lbl * KP;
                    for (int j = 0; j < KP; ++j) a[j] = nc * r * cl[j] * expf(L[j] * INV_EPS);
                }
            }
            for (int j = 0; j < KP; ++j) Ash[t * KP + j] = a[j];
        }
        __syncthreads();
        int lim = min(64, NP - base);
        for (int rr = 0; rr < lim; ++rr) {
            int i = b * NP + base + rr;
            float x = bf2f(Xn[(size_t)i * DIM + t]);
            for (int j = 0; j < KP; ++j) acc[j] += Ash[rr * KP + j] * x;
        }
    }
    for (int j = 0; j < KP; ++j)
        atomicAdd(&Pacc[((size_t)lbl * KP + j) * DIM + t], acc[j]);
}

// ---------------- EMA prototype update ----------------
__global__ void k_ema(const float* __restrict__ proto, const float* __restrict__ Pacc,
                      const int* __restrict__ present, float* __restrict__ o_proto) {
    int i = blockIdx.x * 256 + threadIdx.x;
    if (i >= NC * KP * DIM) return;
    int c = i / (KP * DIM);
    float p = proto[i];
    o_proto[i] = present[c] ? (0.99f * p + 0.01f * Pacc[i]) : p;
}

// ---------------- image_logits = mean over patches ----------------
__global__ __launch_bounds__(256) void k_imglog(const float* __restrict__ logits,
                                                float* __restrict__ o_img) {
    int b = blockIdx.x, t = threadIdx.x;
    float a0 = 0.f, a1 = 0.f, a2 = 0.f, a3 = 0.f;
    const float* base = logits + (size_t)b * NP * COLS;
    for (int n = 0; n < NP; ++n) {
        const float* row = base + (size_t)n * COLS;
        a0 += row[t]; a1 += row[t + 256]; a2 += row[t + 512];
        if (t + 768 < COLS) a3 += row[t + 768];
    }
    const float inv = 1.0f / (float)NP;
    o_img[b * COLS + t] = a0 * inv;
    o_img[b * COLS + t + 256] = a1 * inv;
    o_img[b * COLS + t + 512] = a2 * inv;
    if (t + 768 < COLS) o_img[b * COLS + t + 768] = a3 * inv;
}

// ---------------- class head ----------------
__global__ void k_clslog(const float* __restrict__ o_img, const float* __restrict__ fcw,
                         float* __restrict__ o_cls) {
    int o = blockIdx.x * 256 + threadIdx.x;
    if (o >= BB * NCLS) return;
    int b = o / NCLS, cls = o % NCLS;
    const float* img = o_img + b * COLS;          // first 1000 = classes 0..199
    const float* w = fcw + cls * (NCLS * KP);
    float s = 0.f;
    for (int j = 0; j < NCLS * KP; ++j) s += img[j] * w[j];
    o_cls[o] = s;
}

// ---------------- host launcher ----------------
extern "C" void kernel_launch(void* const* d_in, const int* in_sizes, int n_in,
                              void* d_out, int out_size, void* d_ws, size_t ws_size,
                              hipStream_t stream) {
    const float* X     = (const float*)d_in[0];   // (32,1369,768)
    const float* proto = (const float*)d_in[1];   // (201,5,768)
    const float* fcw   = (const float*)d_in[2];   // (200,1000)
    const int*   labels= (const int*)d_in[3];     // (32,)

    // output regions (concatenated tuple, f32)
    float* out = (float*)d_out;
    float* o_logits = out;
    float* o_img    = o_logits + (size_t)ROWS * COLS;
    float* o_cls    = o_img + BB * COLS;
    float* o_proto  = o_cls + BB * NCLS;
    float* o_pseudo = o_proto + (size_t)NC * KP * DIM;

    // workspace carve (256B aligned)
    char* p = (char*)d_ws;
    auto carve = [&](size_t bytes) -> void* {
        void* r = (void*)p;
        p += (bytes + 255) & ~(size_t)255;
        return r;
    };
    unsigned short* XBUF = (unsigned short*)carve((size_t)ROWS * DIM * 2); // Xt, then Xn
    unsigned short* PNB  = (unsigned short*)carve((size_t)COLS * DIM * 2);
    float* G      = (float*)carve((size_t)DIM * DIM * 4);
    float* mean   = (float*)carve(DIM * 4);
    float* vg     = (float*)carve(DIM * 4);
    float* mv     = (float*)carve(256);
    float* u      = (float*)carve((size_t)ROWS * 4);
    unsigned* mmk = (unsigned*)carve(256);
    int* present  = (int*)carve(NC * 4);
    int* ncnt     = (int*)carve(NC * 4);
    float* S      = (float*)carve(COLS * 4);
    float* cvec   = (float*)carve(COLS * 4);
    float* rbuf   = (float*)carve((size_t)ROWS * 4);
    int* clsbuf   = (int*)carve((size_t)ROWS * 4);
    float* Pacc   = (float*)carve((size_t)NC * KP * DIM * 4);

    k_init<<<3015, 256, 0, stream>>>(Pacc, S, mean, present, ncnt, mmk);

    // PCA path: bf16 transpose -> WMMA Gram -> power iteration -> u -> pseudo labels
    k_transpose<<<1369 * 24, 256, 0, stream>>>(X, XBUF);
    k_gram<<<36, 256, 0, stream>>>(XBUF, G);
    k_mean<<<64, 256, 0, stream>>>(X, mean);
    k_power<<<1, 768, 0, stream>>>(G, mean, vg, mv);
    k_u<<<5476, 256, 0, stream>>>(X, vg, mv, u, mmk);
    k_pseudo<<<32, 256, 0, stream>>>(u, mmk, labels, clsbuf, o_pseudo, rbuf, ncnt, present);

    // main logits GEMM (XBUF reused: Xt no longer needed after k_gram)
    k_norm_tokens<<<5476, 256, 0, stream>>>(X, XBUF);
    k_norm_proto<<<126, 256, 0, stream>>>(proto, PNB);
    k_gemm_logits<<<2738, 256, 0, stream>>>(XBUF, PNB, o_logits);

    // sinkhorn: 3 iterations of (column scale, row scale)
    for (int it = 0; it < 3; ++it) {
        k_skS<<<32, 256, 0, stream>>>(o_logits, labels, rbuf, S);
        k_skC<<<4, 256, 0, stream>>>(S, cvec);
        k_skR<<<172, 256, 0, stream>>>(o_logits, clsbuf, cvec, ncnt, rbuf);
    }
    k_paccum<<<32, 768, 0, stream>>>(o_logits, XBUF, labels, rbuf, cvec, ncnt, Pacc);
    k_ema<<<3015, 256, 0, stream>>>(proto, Pacc, present, o_proto);

    // heads
    k_imglog<<<32, 256, 0, stream>>>(o_logits, o_img);
    k_clslog<<<25, 256, 0, stream>>>(o_img, fcw, o_cls);
}
